// Block_27685359190357
// MI455X (gfx1250) — matrix operand order repaired and checked by reading, not compiled, not run on hardware
//
#include <hip/hip_runtime.h>
#include <hip/hip_bf16.h>

typedef unsigned short u16;
typedef unsigned int   u32;
typedef __attribute__((ext_vector_type(16))) __bf16 v16bf;
typedef __attribute__((ext_vector_type(8)))  float  v8f;

#define B_  2
#define S_  2048
#define D_  1024
#define H_  16
#define HD_ 64
#define N_  (B_*S_)     // 4096 tokens
#define E_  8
#define HM_ 1024
#define HS_ 1024
#define LDA 40          // A-tile LDS stride (halves): 80B, 16B-aligned rows, conflict-free frag reads
#define LDB 34          // B-tile LDS stride (halves): 68B = 17 dwords, gcd(17,64)=1
#define LDT 34          // flash-attn P stride
#define AT_LDK 66       // flash-attn K/V stride: 132B = 33 dwords, gcd(33,64)=1

#if defined(__has_builtin)
#if __has_builtin(__builtin_amdgcn_global_load_async_to_lds_b128) && \
    __has_builtin(__builtin_amdgcn_s_wait_asynccnt)
#define USE_ASYNC 1
#endif
#endif

#define GAS __attribute__((address_space(1)))
#define LAS __attribute__((address_space(3)))

typedef __attribute__((__vector_size__(4 * sizeof(int)))) int i128;   // b128 payload type

union FragAB { u32 w[8]; v16bf v; };

static __device__ __forceinline__ u16 f2bf(float f) {
    u32 u = __float_as_uint(f);
    u32 r = (u + 0x7FFFu + ((u >> 16) & 1u)) >> 16;   // round-to-nearest-even
    return (u16)r;
}

// ---------------------------------------------------------------- cast f32->bf16
__global__ void cast_bf16_kernel(const float* __restrict__ src, u16* __restrict__ dst, int n) {
    int i = blockIdx.x * blockDim.x + threadIdx.x;
    if (i < n) dst[i] = f2bf(src[i]);
}

// ---------------------------------------------------------------- RMSNorm (row per block)
__global__ __launch_bounds__(256) void rmsnorm_kernel(const float* __restrict__ x,
                                                      const float* __restrict__ w,
                                                      u16* __restrict__ outb,
                                                      float* __restrict__ outf) {
    __shared__ float red[8];
    const int row = blockIdx.x, tid = threadIdx.x;
    const float* xr = x + (size_t)row * D_;
    float ss = 0.f;
    for (int j = tid; j < D_; j += 256) { float v = xr[j]; ss += v * v; }
    for (int off = 16; off; off >>= 1) ss += __shfl_xor(ss, off, 32);
    if ((tid & 31) == 0) red[tid >> 5] = ss;
    __syncthreads();
    if (tid == 0) { float t = 0.f; for (int i = 0; i < 8; ++i) t += red[i]; red[0] = t; }
    __syncthreads();
    const float scale = rsqrtf(red[0] / (float)D_ + 1e-6f);
    for (int j = tid; j < D_; j += 256) {
        float v = xr[j] * scale * w[j];
        outb[(size_t)row * D_ + j] = f2bf(v);
        if (outf) outf[(size_t)row * D_ + j] = v;
    }
}

// ---------------------------------------------------------------- WMMA GEMM
// C[M,N] (f32) = A[M,K] (bf16, row-major) * B[K,N] (bf16, row-major)
// block = 256 thr (8 waves); block tile 256x128; wave tile 64x64; K-step 32.
// A tile staged with GLOBAL_LOAD_ASYNC_TO_LDS_B128 when available (ASYNCcnt path).
__global__ __launch_bounds__(256) void gemm_bf16_wmma(const u16* __restrict__ A,
                                                      const u16* __restrict__ B,
                                                      float* __restrict__ C,
                                                      int M, int Nn, int K) {
    __shared__ u16 Alds[256 * LDA];      // [row][k], 16B-aligned rows
    __shared__ u16 Blds[128 * LDB];      // transposed: [col][k], padded
    const int tid  = threadIdx.x;
    const int lane = tid & 31;
    const int wave = tid >> 5;
    const int wmy  = wave >> 1;          // 0..3  -> M offset wmy*64
    const int wnx  = wave & 1;           // 0..1  -> N offset wnx*64
    const int half = (lane < 16) ? 0 : 1;
    const int l16  = lane & 15;
    const int m0 = blockIdx.y * 256;
    const int n0 = blockIdx.x * 128;

    v8f acc[4][4];
    #pragma unroll
    for (int mt = 0; mt < 4; ++mt)
        #pragma unroll
        for (int nt = 0; nt < 4; ++nt)
            #pragma unroll
            for (int i = 0; i < 8; ++i) acc[mt][nt][i] = 0.f;

    for (int kb = 0; kb < K; kb += 32) {
        { // A tile 256x32: thread -> one full 64B row
            const u16* ga = A + (size_t)(m0 + tid) * K + kb;
            u16* la = &Alds[tid * LDA];
            if (kb + 32 < K) __builtin_prefetch(ga + 32, 0, 1);
#ifdef USE_ASYNC
            __builtin_amdgcn_global_load_async_to_lds_b128((GAS i128*)ga, (LAS i128*)la, 0,  0);
            __builtin_amdgcn_global_load_async_to_lds_b128((GAS i128*)ga, (LAS i128*)la, 16, 0);
            __builtin_amdgcn_global_load_async_to_lds_b128((GAS i128*)ga, (LAS i128*)la, 32, 0);
            __builtin_amdgcn_global_load_async_to_lds_b128((GAS i128*)ga, (LAS i128*)la, 48, 0);
#else
            const u32* g32 = (const u32*)ga;
            u32* l32 = (u32*)la;
            #pragma unroll
            for (int j = 0; j < 16; ++j) l32[j] = g32[j];
#endif
        }
        { // B tile 32x128 -> LDS transposed [col][k]
            int krow = tid & 31, cb = (tid >> 5) * 16;
            const u16* g = B + (size_t)(kb + krow) * Nn + n0 + cb;
            if (kb + 32 < K) __builtin_prefetch(B + (size_t)(kb + 32 + krow) * Nn + n0 + cb, 0, 1);
            #pragma unroll
            for (int j = 0; j < 16; ++j) Blds[(cb + j) * LDB + krow] = g[j];
        }
#ifdef USE_ASYNC
        __builtin_amdgcn_s_wait_asynccnt(0);
#endif
        __syncthreads();

        FragAB afr[4], bfr[4];
        #pragma unroll
        for (int mt = 0; mt < 4; ++mt) {      // A 16x32 layout: lane<16 k=0..15, lane>=16 k=16..31
            int arow = wmy * 64 + mt * 16 + l16;
            int kba  = half * 16;
            const u32* s0 = (const u32*)&Alds[arow * LDA + kba];
            const u32* s1 = (const u32*)&Alds[arow * LDA + kba + 8];
            #pragma unroll
            for (int j = 0; j < 4; ++j) { afr[mt].w[j] = s0[j]; afr[mt].w[4 + j] = s1[j]; }
        }
        #pragma unroll
        for (int nt = 0; nt < 4; ++nt) {      // B 32x16 layout: lane<16 k=0..15, lane>=16 k=16..31
            int col = wnx * 64 + nt * 16 + l16;
            const u32* s0 = (const u32*)&Blds[col * LDB + half * 16];
            #pragma unroll
            for (int j = 0; j < 8; ++j) bfr[nt].w[j] = s0[j];
        }
        #pragma unroll
        for (int mt = 0; mt < 4; ++mt)
            #pragma unroll
            for (int nt = 0; nt < 4; ++nt)
                acc[mt][nt] = __builtin_amdgcn_wmma_f32_16x16x32_bf16(
                    false, afr[mt].v, false, bfr[nt].v, (short)0, acc[mt][nt], false, false);
        __syncthreads();
    }
    #pragma unroll
    for (int mt = 0; mt < 4; ++mt)
        #pragma unroll
        for (int nt = 0; nt < 4; ++nt)
            #pragma unroll
            for (int i = 0; i < 8; ++i) {
                int row = m0 + wmy * 64 + mt * 16 + half * 8 + i;
                int col = n0 + wnx * 64 + nt * 16 + l16;
                C[(size_t)row * Nn + col] = acc[mt][nt][i];
            }
}

// ---------------------------------------------------------------- RoPE + [B,H,S,HD] reorder
__global__ void rope_reorder_kernel(const float* __restrict__ src, const float* __restrict__ cs,
                                    const float* __restrict__ sn, u16* __restrict__ dst) {
    int idx = blockIdx.x * blockDim.x + threadIdx.x;       // N*H*32 rotation pairs
    if (idx >= N_ * H_ * 32) return;
    int i = idx & 31, hh = (idx >> 5) & (H_ - 1), tok = idx >> 9;
    int b = tok / S_, s = tok % S_;
    float t0 = src[(size_t)tok * D_ + hh * HD_ + 2 * i];
    float t1 = src[(size_t)tok * D_ + hh * HD_ + 2 * i + 1];
    float c = cs[s * 32 + i], sv = sn[s * 32 + i];
    size_t db = ((size_t)(b * H_ + hh) * S_ + s) * HD_;
    dst[db + 2 * i]     = f2bf(t0 * c - t1 * sv);
    dst[db + 2 * i + 1] = f2bf(t0 * sv + t1 * c);
}

__global__ void v_reorder_kernel(const float* __restrict__ src, u16* __restrict__ dst) {
    int idx = blockIdx.x * blockDim.x + threadIdx.x;       // N*D
    if (idx >= N_ * D_) return;
    int tok = idx / D_, d = idx % D_;
    int hh = d / HD_, hd = d % HD_;
    int b = tok / S_, s = tok % S_;
    dst[((size_t)(b * H_ + hh) * S_ + s) * HD_ + hd] = f2bf(src[idx]);
}

// ---------------------------------------------------------------- Flash attention (WMMA)
// Q,K,V bf16 [BH][S][64]; out bf16 [N][D] (token-major, col = h*64+hd)
__global__ __launch_bounds__(128) void flash_attn_wmma(const u16* __restrict__ Q,
                                                       const u16* __restrict__ Km,
                                                       const u16* __restrict__ Vm,
                                                       u16* __restrict__ O) {
    __shared__ u16 Klds[32 * AT_LDK];
    __shared__ u16 Vlds[32 * AT_LDK];
    __shared__ u16 Plds[4 * 16 * LDT];
    const int bh = blockIdx.x;                 // 0..31
    const int q0 = blockIdx.y * 64;            // 64 q rows per block
    const int tid = threadIdx.x, lane = tid & 31, w = tid >> 5;
    const int half = (lane < 16) ? 0 : 1, l16 = lane & 15;
    const size_t base = (size_t)bh * S_ * HD_;

    // Q tile (16 rows) -> two A fragments (chan 0..31, 32..63), straight from global
    FragAB qfr[2];
    {
        const u16* qp = Q + base + (size_t)(q0 + w * 16 + l16) * HD_;
        #pragma unroll
        for (int ks = 0; ks < 2; ++ks) {
            int kb = ks * 32 + half * 16;
            const u32* g0 = (const u32*)(qp + kb);
            const u32* g1 = (const u32*)(qp + kb + 8);
            #pragma unroll
            for (int j = 0; j < 4; ++j) { qfr[ks].w[j] = g0[j]; qfr[ks].w[4 + j] = g1[j]; }
        }
    }

    float mrow[8], lrow[8];
    v8f o[4];
    #pragma unroll
    for (int i = 0; i < 8; ++i) { mrow[i] = -1e30f; lrow[i] = 0.f; }
    #pragma unroll
    for (int ct = 0; ct < 4; ++ct)
        #pragma unroll
        for (int i = 0; i < 8; ++i) o[ct][i] = 0.f;

    for (int kc = 0; kc < S_ / 32; ++kc) {
        __syncthreads();
        { // cooperative K/V chunk load: 32 keys x 64 chans each
            int key = tid >> 2, seg = (tid & 3) * 16;
            const u32* kg = (const u32*)(Km + base + (size_t)(kc * 32 + key) * HD_ + seg);
            const u32* vg = (const u32*)(Vm + base + (size_t)(kc * 32 + key) * HD_ + seg);
            u32* ks_ = (u32*)&Klds[key * AT_LDK + seg];
            u32* vs_ = (u32*)&Vlds[key * AT_LDK + seg];
            #pragma unroll
            for (int j = 0; j < 8; ++j) { ks_[j] = kg[j]; vs_[j] = vg[j]; }
        }
        __syncthreads();

        // scores: two 16x16 key tiles, K-dim = channels (2 steps of 32)
        v8f s[2];
        #pragma unroll
        for (int kt = 0; kt < 2; ++kt)
            #pragma unroll
            for (int i = 0; i < 8; ++i) s[kt][i] = 0.f;
        #pragma unroll
        for (int cstep = 0; cstep < 2; ++cstep) {
            #pragma unroll
            for (int kt = 0; kt < 2; ++kt) {
                FragAB kf;   // B frag of K^T: lane=key col, elements = channels
                const u32* kp = (const u32*)&Klds[(kt * 16 + l16) * AT_LDK + cstep * 32 + half * 16];
                #pragma unroll
                for (int j = 0; j < 8; ++j) kf.w[j] = kp[j];
                s[kt] = __builtin_amdgcn_wmma_f32_16x16x32_bf16(
                    false, qfr[cstep].v, false, kf.v, (short)0, s[kt], false, false);
            }
        }

        // online softmax over this 32-key chunk (row = half*8 + i, col = lane%16)
        #pragma unroll
        for (int i = 0; i < 8; ++i) {
            float s0 = s[0][i] * 0.125f, s1 = s[1][i] * 0.125f;
            float v = fmaxf(s0, s1);
            for (int off = 1; off < 16; off <<= 1) v = fmaxf(v, __shfl_xor(v, off, 32));
            float mn = fmaxf(mrow[i], v);
            float al = __expf(mrow[i] - mn);
            mrow[i] = mn;
            float p0 = __expf(s0 - mn), p1 = __expf(s1 - mn);
            float rs = p0 + p1;
            for (int off = 1; off < 16; off <<= 1) rs += __shfl_xor(rs, off, 32);
            lrow[i] = lrow[i] * al + rs;
            #pragma unroll
            for (int ct = 0; ct < 4; ++ct) o[ct][i] *= al;
            int prow = w * 16 + half * 8 + i;
            Plds[prow * LDT + l16]      = f2bf(p0);
            Plds[prow * LDT + 16 + l16] = f2bf(p1);
        }

        // P (16x32) as A fragment (per-wave LDS region; same-wave LDS is in-order)
        FragAB pfr;
        {
            const u32* p0 = (const u32*)&Plds[(w * 16 + l16) * LDT + half * 16];
            const u32* p1 = (const u32*)&Plds[(w * 16 + l16) * LDT + half * 16 + 8];
            #pragma unroll
            for (int j = 0; j < 4; ++j) { pfr.w[j] = p0[j]; pfr.w[4 + j] = p1[j]; }
        }
        // O += P * V  (4 channel tiles)
        #pragma unroll
        for (int ct = 0; ct < 4; ++ct) {
            FragAB vf;   // B frag of V: lane = channel col, elements = keys
            int chan = ct * 16 + l16, keyb = half * 16;
            #pragma unroll
            for (int j = 0; j < 8; ++j) {
                u16 lo = Vlds[(keyb + 2 * j)     * AT_LDK + chan];
                u16 hi = Vlds[(keyb + 2 * j + 1) * AT_LDK + chan];
                vf.w[j] = (u32)lo | ((u32)hi << 16);
            }
            o[ct] = __builtin_amdgcn_wmma_f32_16x16x32_bf16(
                false, pfr.v, false, vf.v, (short)0, o[ct], false, false);
        }
    }

    // finalize: O /= l, write to [token][h*64+hd]
    const int b = bh / H_, hh = bh % H_;
    #pragma unroll
    for (int i = 0; i < 8; ++i) {
        float inv = 1.f / lrow[i];
        int srow = q0 + w * 16 + half * 8 + i;
        size_t orow = (size_t)(b * S_ + srow) * D_ + hh * HD_;
        #pragma unroll
        for (int ct = 0; ct < 4; ++ct)
            O[orow + ct * 16 + l16] = f2bf(o[ct][i] * inv);
    }
}

// ---------------------------------------------------------------- gate softmax + top-2
__global__ void gate_topk_kernel(const float* __restrict__ xf, const float* __restrict__ gw,
                                 float* __restrict__ mask) {
    int n = blockIdx.x * blockDim.x + threadIdx.x;
    if (n >= N_) return;
    float sc[E_];
    #pragma unroll
    for (int e = 0; e < E_; ++e) sc[e] = 0.f;
    const float* xr = xf + (size_t)n * D_;
    for (int j = 0; j < D_; ++j) {
        float xv = xr[j];
        #pragma unroll
        for (int e = 0; e < E_; ++e) sc[e] += xv * gw[e * D_ + j];
    }
    float mx = sc[0];
    #pragma unroll
    for (int e = 1; e < E_; ++e) mx = fmaxf(mx, sc[e]);
    float sum = 0.f, p[E_];
    #pragma unroll
    for (int e = 0; e < E_; ++e) { p[e] = __expf(sc[e] - mx); sum += p[e]; }
    #pragma unroll
    for (int e = 0; e < E_; ++e) p[e] /= sum;
    int a0 = 0;
    #pragma unroll
    for (int e = 1; e < E_; ++e) if (p[e] > p[a0]) a0 = e;
    int a1 = (a0 == 0) ? 1 : 0;
    #pragma unroll
    for (int e = 0; e < E_; ++e) if (e != a0 && p[e] > p[a1]) a1 = e;
    float s2 = p[a0] + p[a1] + 1e-9f;
    #pragma unroll
    for (int e = 0; e < E_; ++e) mask[(size_t)n * E_ + e] = 0.f;
    mask[(size_t)n * E_ + a0] = p[a0] / s2;
    mask[(size_t)n * E_ + a1] = p[a1] / s2;
}

// ---------------------------------------------------------------- elementwise helpers
__global__ void silu_mul_kernel(const float* __restrict__ a, const float* __restrict__ c,
                                u16* __restrict__ out, int n) {
    int i = blockIdx.x * blockDim.x + threadIdx.x;
    if (i >= n) return;
    float av = a[i];
    out[i] = f2bf((av / (1.f + __expf(-av))) * c[i]);
}

__global__ void add2_kernel(const float* __restrict__ a, const float* __restrict__ b,
                            float* __restrict__ out, int n) {
    int i = blockIdx.x * blockDim.x + threadIdx.x;
    if (i < n) out[i] = a[i] + b[i];
}

__global__ void axpy_mask_kernel(float* __restrict__ y, const float* __restrict__ eo,
                                 const float* __restrict__ mask, int e, int n) {
    int i = blockIdx.x * blockDim.x + threadIdx.x;
    if (i >= n) return;
    float wgt = mask[(size_t)(i / D_) * E_ + e];
    y[i] += wgt * eo[i];
}

// ---------------------------------------------------------------- host orchestration
static void launch_gemm(const u16* A, const u16* B, float* C, int M, int Nn, int K,
                        hipStream_t s) {
    dim3 g(Nn / 128, M / 256);
    gemm_bf16_wmma<<<g, 256, 0, s>>>(A, B, C, M, Nn, K);
}

extern "C" void kernel_launch(void* const* d_in, const int* in_sizes, int n_in,
                              void* d_out, int out_size, void* d_ws, size_t ws_size,
                              hipStream_t stream) {
    (void)in_sizes; (void)n_in; (void)out_size; (void)ws_size;
    const float* x    = (const float*)d_in[0];
    const float* fcos = (const float*)d_in[1];
    const float* fsin = (const float*)d_in[2];
    const float* anw  = (const float*)d_in[3];
    const float* fnw  = (const float*)d_in[4];
    const float* wq   = (const float*)d_in[5];
    const float* wk   = (const float*)d_in[6];
    const float* wv   = (const float*)d_in[7];
    const float* wo   = (const float*)d_in[8];
    const float* gw   = (const float*)d_in[9];
    const float* ew1  = (const float*)d_in[10];
    const float* ew2  = (const float*)d_in[11];
    const float* ew3  = (const float*)d_in[12];
    const float* sw1  = (const float*)d_in[13];
    const float* sw2  = (const float*)d_in[14];
    const float* sw3  = (const float*)d_in[15];
    float* out = (float*)d_out;

    char* p = (char*)d_ws;
    auto alloc = [&](size_t bytes) { void* r = p; p += (bytes + 255) & ~(size_t)255; return r; };
    const size_t ND = (size_t)N_ * D_;
    u16* wqb  = (u16*)alloc((size_t)D_ * D_ * 2);
    u16* wkb  = (u16*)alloc((size_t)D_ * D_ * 2);
    u16* wvb  = (u16*)alloc((size_t)D_ * D_ * 2);
    u16* wob  = (u16*)alloc((size_t)D_ * D_ * 2);
    u16* sw1b = (u16*)alloc((size_t)D_ * HS_ * 2);
    u16* sw2b = (u16*)alloc((size_t)HS_ * D_ * 2);
    u16* sw3b = (u16*)alloc((size_t)D_ * HS_ * 2);
    u16* ew1b = (u16*)alloc((size_t)E_ * D_ * HM_ * 2);
    u16* ew2b = (u16*)alloc((size_t)E_ * HM_ * D_ * 2);
    u16* ew3b = (u16*)alloc((size_t)E_ * D_ * HM_ * 2);
    u16* xnb  = (u16*)alloc(ND * 2);
    u16* qbr  = (u16*)alloc(ND * 2);
    u16* kbr  = (u16*)alloc(ND * 2);
    u16* vbr  = (u16*)alloc(ND * 2);
    u16* aob  = (u16*)alloc(ND * 2);
    u16* xfb  = (u16*)alloc(ND * 2);
    u16* h1b  = (u16*)alloc((size_t)N_ * HM_ * 2);
    float* qf   = (float*)alloc(ND * 4);
    float* kf   = (float*)alloc(ND * 4);
    float* vf   = (float*)alloc(ND * 4);
    float* hres = (float*)alloc(ND * 4);
    float* xff  = (float*)alloc(ND * 4);
    float* t1   = (float*)alloc((size_t)N_ * HM_ * 4);
    float* t3   = (float*)alloc((size_t)N_ * HM_ * 4);
    float* eo   = (float*)alloc(ND * 4);
    float* yacc = (float*)alloc(ND * 4);
    float* maskb = (float*)alloc((size_t)N_ * E_ * 4);

    auto cast = [&](const float* s, u16* d, int n) {
        cast_bf16_kernel<<<(n + 255) / 256, 256, 0, stream>>>(s, d, n);
    };
    cast(wq, wqb, D_ * D_);  cast(wk, wkb, D_ * D_);
    cast(wv, wvb, D_ * D_);  cast(wo, wob, D_ * D_);
    cast(sw1, sw1b, D_ * HS_); cast(sw2, sw2b, HS_ * D_); cast(sw3, sw3b, D_ * HS_);
    cast(ew1, ew1b, E_ * D_ * HM_); cast(ew2, ew2b, E_ * HM_ * D_); cast(ew3, ew3b, E_ * D_ * HM_);

    // ---- attention half
    rmsnorm_kernel<<<N_, 256, 0, stream>>>(x, anw, xnb, nullptr);
    launch_gemm(xnb, wqb, qf, N_, D_, D_, stream);
    launch_gemm(xnb, wkb, kf, N_, D_, D_, stream);
    launch_gemm(xnb, wvb, vf, N_, D_, D_, stream);
    {
        int n = N_ * H_ * 32;
        rope_reorder_kernel<<<(n + 255) / 256, 256, 0, stream>>>(qf, fcos, fsin, qbr);
        rope_reorder_kernel<<<(n + 255) / 256, 256, 0, stream>>>(kf, fcos, fsin, kbr);
        v_reorder_kernel<<<((int)ND + 255) / 256, 256, 0, stream>>>(vf, vbr);
    }
    flash_attn_wmma<<<dim3(B_ * H_, S_ / 64), 128, 0, stream>>>(qbr, kbr, vbr, aob);
    launch_gemm(aob, wob, qf /*reuse as attn-proj*/, N_, D_, D_, stream);
    add2_kernel<<<((int)ND + 255) / 256, 256, 0, stream>>>(x, qf, hres, (int)ND);

    // ---- FFN half
    rmsnorm_kernel<<<N_, 256, 0, stream>>>(hres, fnw, xfb, xff);
    gate_topk_kernel<<<(N_ + 63) / 64, 64, 0, stream>>>(xff, gw, maskb);

    // shared expert -> yacc
    launch_gemm(xfb, sw1b, t1, N_, HS_, D_, stream);
    launch_gemm(xfb, sw3b, t3, N_, HS_, D_, stream);
    silu_mul_kernel<<<(N_ * HS_ + 255) / 256, 256, 0, stream>>>(t1, t3, h1b, N_ * HS_);
    launch_gemm(h1b, sw2b, yacc, N_, D_, HS_, stream);

    // routed experts (dense, matching reference semantics), masked accumulate
    for (int e = 0; e < E_; ++e) {
        launch_gemm(xfb, ew1b + (size_t)e * D_ * HM_, t1, N_, HM_, D_, stream);
        launch_gemm(xfb, ew3b + (size_t)e * D_ * HM_, t3, N_, HM_, D_, stream);
        silu_mul_kernel<<<(N_ * HM_ + 255) / 256, 256, 0, stream>>>(t1, t3, h1b, N_ * HM_);
        launch_gemm(h1b, ew2b + (size_t)e * HM_ * D_, eo, N_, D_, HM_, stream);
        axpy_mask_kernel<<<((int)ND + 255) / 256, 256, 0, stream>>>(yacc, eo, maskb, e, (int)ND);
    }

    add2_kernel<<<((int)ND + 255) / 256, 256, 0, stream>>>(hres, yacc, out, (int)ND);
}
